// LSTMAutoEncoder_81990925680856
// MI455X (gfx1250) — matrix operand order repaired and checked by reading, not compiled, not run on hardware
//
#include <hip/hip_runtime.h>
#include <hip/hip_bf16.h>

// ---------------- problem constants ----------------
#define NWG 16       // persistent workgroups (16 resident -> barrier-safe)
#define BB  256      // batch
#define SS  512      // sequence length
#define DD  64       // input / output dim
#define HH  128      // hidden dim
#define BH  (BB*HH)  // 32768 elements of h state

typedef __attribute__((ext_vector_type(16))) _Float16 v16h;
typedef __attribute__((ext_vector_type(8)))  float    v8f;

union AB32 { v16h v; uint4 q[2]; };

// ---- WMMA fragment loads (16x16x32 f16, wave32 layouts per CDNA5 ISA 7.12.2) ----
// A (16x32, M x K): lane l<16 -> M=l, VGPR v: K = 2*(v&3) + (v>=4?16:0) (+8 for lanes 16-31)
// => per lane two contiguous 16B chunks at halfs [k0+laneHi*8, +8) and [k0+16+laneHi*8, +8)
__device__ __forceinline__ v16h load_afrag(const _Float16* rowbase, int k0, int laneHi) {
    AB32 u;
    const _Float16* p = rowbase + k0 + laneHi * 8;
    u.q[0] = *(const uint4*)(p);
    u.q[1] = *(const uint4*)(p + 16);
    return u.v;
}
// B (32x16, K x N): lane l -> N=l&15; same K order as A: halfs [k0+laneHi*8, +8) and [k0+16+laneHi*8, +8)
// weight slice stored row-major [n][K] => two 16B chunks per lane
__device__ __forceinline__ v16h load_bfrag(const _Float16* rowbase, int k0, int laneHi) {
    AB32 u;
    const _Float16* p = rowbase + k0 + laneHi * 8;
    u.q[0] = *(const uint4*)(p);
    u.q[1] = *(const uint4*)(p + 16);
    return u.v;
}

#define WMMA(A, B, C) __builtin_amdgcn_wmma_f32_16x16x32_f16(false, (A), false, (B), (short)0, (C), false, false)

__device__ __forceinline__ float sigm(float x)  { return 1.0f / (1.0f + __expf(-x)); }
__device__ __forceinline__ float tanh_(float x) { return 1.0f - 2.0f / (__expf(2.0f * x) + 1.0f); }

// ---------------- global (grid) barrier among NWG persistent workgroups ----------------
__device__ __forceinline__ void gridbar(unsigned* cnt, unsigned& target) {
    __threadfence();          // make h stores visible at agent scope
    __syncthreads();
    if (threadIdx.x == 0) {
        __hip_atomic_fetch_add(cnt, 1u, __ATOMIC_RELEASE, __HIP_MEMORY_SCOPE_AGENT);
        while (__hip_atomic_load(cnt, __ATOMIC_ACQUIRE, __HIP_MEMORY_SCOPE_AGENT) < target) {
            __builtin_amdgcn_s_sleep(2);
        }
        __builtin_amdgcn_fence(__ATOMIC_ACQUIRE, "agent"); // invalidate WGP$ for stale h
    }
    __syncthreads();
    target += NWG;
}

// stage this block's 64 gate-columns (hidden group hg) of combined weight [512][Kpad] into LDS
__device__ __forceinline__ void stage_weights(const _Float16* __restrict__ Wc,
                                              _Float16* lds, int Kpad, int hg) {
    const int total = 64 * Kpad;                 // Kpad is a multiple of 8 -> uint4 chunks never cross rows
    for (int i = threadIdx.x * 8; i < total; i += blockDim.x * 8) {
        int row  = i / Kpad;                      // 0..63 = gate*16 + r
        int k    = i - row * Kpad;
        int gate = row >> 4, r = row & 15;
        *(uint4*)(lds + i) = *(const uint4*)(Wc + (size_t)(gate * HH + hg * 16 + r) * Kpad + k);
    }
}

// ---------------- one LSTM cell step for this wave's (m_tile, hg) 16x16 tile ----------------
template<int KIN, int KPAD>
__device__ __forceinline__ void cell_step(
    int m_tile, int hg, int ln, int laneHi,
    const _Float16* __restrict__ xin, int xstride,   // nullable (skip input GEMM)
    const _Float16* __restrict__ hin,                // nullable (h_prev == 0), stride HH
    const _Float16* wlds,                            // LDS [64][KPAD]
    const float* __restrict__ bias,                  // [512] (bih+bhh)
    v8f& ct,                                         // persistent c tile (C/D layout)
    _Float16* __restrict__ hout)                     // [BB][HH] f16
{
    v8f a0, a1, a2, a3;
    {
        const int nb = hg * 16 + ln;
        float b0 = bias[0 * HH + nb], b1 = bias[1 * HH + nb];
        float b2 = bias[2 * HH + nb], b3 = bias[3 * HH + nb];
#pragma unroll
        for (int r = 0; r < 8; ++r) { a0[r] = b0; a1[r] = b1; a2[r] = b2; a3[r] = b3; }
    }
    const int brow = m_tile * 16 + ln;
    const _Float16* w0 = wlds + (0 * 16 + ln) * KPAD;
    const _Float16* w1 = wlds + (1 * 16 + ln) * KPAD;
    const _Float16* w2 = wlds + (2 * 16 + ln) * KPAD;
    const _Float16* w3 = wlds + (3 * 16 + ln) * KPAD;

    if (xin) {
        const _Float16* ar = xin + (size_t)brow * xstride;
#pragma unroll
        for (int k0 = 0; k0 < KIN; k0 += 32) {
            v16h a = load_afrag(ar, k0, laneHi);
            a0 = WMMA(a, load_bfrag(w0, k0, laneHi), a0);
            a1 = WMMA(a, load_bfrag(w1, k0, laneHi), a1);
            a2 = WMMA(a, load_bfrag(w2, k0, laneHi), a2);
            a3 = WMMA(a, load_bfrag(w3, k0, laneHi), a3);
        }
    }
    if (hin) {
        const _Float16* ar = hin + (size_t)brow * HH;
#pragma unroll
        for (int k0 = 0; k0 < HH; k0 += 32) {
            v16h a = load_afrag(ar, k0, laneHi);
            a0 = WMMA(a, load_bfrag(w0, KIN + k0, laneHi), a0);
            a1 = WMMA(a, load_bfrag(w1, KIN + k0, laneHi), a1);
            a2 = WMMA(a, load_bfrag(w2, KIN + k0, laneHi), a2);
            a3 = WMMA(a, load_bfrag(w3, KIN + k0, laneHi), a3);
        }
    }
    // gates (i,f,g,o) -> c,h   (each lane: 8 elements of the 16x16 tile, C/D layout)
    v8f hv;
#pragma unroll
    for (int r = 0; r < 8; ++r) {
        float iv = sigm(a0[r]);
        float fv = sigm(a1[r]);
        float gv = tanh_(a2[r]);
        float ov = sigm(a3[r]);
        float cc = fv * ct[r] + iv * gv;
        ct[r] = cc;
        hv[r] = ov * tanh_(cc);
    }
    const int col = hg * 16 + ln;
#pragma unroll
    for (int r = 0; r < 8; ++r)
        hout[(size_t)(m_tile * 16 + laneHi * 8 + r) * HH + col] = (_Float16)hv[r];
}

// ---------------- persistent autoencoder kernel ----------------
__launch_bounds__(256, 1)
__global__ void lstm_ae_persistent(
    const _Float16* __restrict__ xh,     // [BB][SS][DD] f16
    _Float16* __restrict__ y0,           // [SS][BB][HH] f16 (enc layer0 outputs)
    _Float16* __restrict__ h0buf,        // [2][BB][HH]
    _Float16* __restrict__ h1buf,        // [2][BB][HH]
    _Float16* __restrict__ predb,        // [2][BB][DD]
    const _Float16* __restrict__ w_enc0, const _Float16* __restrict__ w_enc1,
    const _Float16* __restrict__ w_dec0, const _Float16* __restrict__ w_dec1,
    const float* __restrict__ b_enc0, const float* __restrict__ b_enc1,
    const float* __restrict__ b_dec0, const float* __restrict__ b_dec1,
    const _Float16* __restrict__ fcWh,   // [DD][HH] f16
    const float* __restrict__ fcb,       // [DD] f32
    float* __restrict__ out,             // [BB][SS][DD] f32
    unsigned* __restrict__ bar)
{
    __shared__ __align__(16) _Float16 wlds0[64 * 192];   // 24 KB
    __shared__ __align__(16) _Float16 wlds1[64 * 256];   // 32 KB

    const int tid    = threadIdx.x;
    const int wave   = tid >> 5;
    const int lane   = tid & 31;
    const int ln     = lane & 15;
    const int laneHi = lane >> 4;
    const int hg     = blockIdx.x & 7;                 // hidden group 0..7 (16 cols each)
    const int m_tile = (blockIdx.x >> 3) * 8 + wave;   // 0..15 (16 batch rows each)
    const int wgid   = blockIdx.x * 8 + wave;          // 0..127

    unsigned target = NWG;
    v8f c0, c1;
#pragma unroll
    for (int r = 0; r < 8; ++r) { c0[r] = 0.0f; c1[r] = 0.0f; }

    // ================= encoder =================
    stage_weights(w_enc0, wlds0, 192, hg);
    stage_weights(w_enc1, wlds1, 256, hg);
    __syncthreads();

    // layer 0: x_t (K=64) + h_{t-1} (K=128); h stream stored to y0[t]
    for (int t = 0; t < SS; ++t) {
        cell_step<64, 192>(m_tile, hg, ln, laneHi,
                           xh + (size_t)t * DD, SS * DD,
                           t ? (y0 + (size_t)(t - 1) * BH) : nullptr,
                           wlds0, b_enc0, c0, y0 + (size_t)t * BH);
        gridbar(bar, target);
    }
    // layer 1: y0[t] (K=128) + h_{t-1} (K=128); double-buffered h1
    for (int t = 0; t < SS; ++t) {
        cell_step<128, 256>(m_tile, hg, ln, laneHi,
                            y0 + (size_t)t * BH, HH,
                            t ? (h1buf + (size_t)((t - 1) & 1) * BH) : nullptr,
                            wlds1, b_enc1, c1, h1buf + (size_t)(t & 1) * BH);
        gridbar(bar, target);
    }

    // ================= decoder =================
    __syncthreads();
    stage_weights(w_dec0, wlds0, 192, hg);
    stage_weights(w_dec1, wlds1, 256, hg);
    __syncthreads();

    // FC B-fragments held in registers by waves 0..63 (tile = (wgid>>2, wgid&3))
    v16h fcB[4];
    if (wgid < 64) {
        const int nt = wgid & 3;
#pragma unroll
        for (int kc = 0; kc < 4; ++kc)
            fcB[kc] = load_bfrag(fcWh + (size_t)(nt * 16 + ln) * HH, kc * 32, laneHi);
    }

    for (int t = 0; t < SS; ++t) {
        // cell0: input = pred_{t-1} (zeros at t=0), h from enc-l0 final at t=0
        cell_step<64, 192>(m_tile, hg, ln, laneHi,
                           t ? (predb + (size_t)((t - 1) & 1) * BB * DD) : nullptr, DD,
                           t ? (h0buf + (size_t)((t - 1) & 1) * BH) : (y0 + (size_t)(SS - 1) * BH),
                           wlds0, b_dec0, c0, h0buf + (size_t)(t & 1) * BH);
        gridbar(bar, target);

        // cell1: input = h0_t ; h1 carries over from encoder final (parity 1 at t=0)
        cell_step<128, 256>(m_tile, hg, ln, laneHi,
                            h0buf + (size_t)(t & 1) * BH, HH,
                            h1buf + (size_t)((t - 1) & 1) * BH,
                            wlds1, b_dec1, c1, h1buf + (size_t)(t & 1) * BH);
        gridbar(bar, target);

        // FC: pred = h1_t @ fcW^T + fcb ; 64 tiles on waves 0..63
        if (wgid < 64) {
            const int mt = wgid >> 2, nt = wgid & 3;
            const int col = nt * 16 + ln;
            v8f acc; float bv = fcb[col];
#pragma unroll
            for (int r = 0; r < 8; ++r) acc[r] = bv;
            const _Float16* ar = h1buf + (size_t)(t & 1) * BH + (size_t)(mt * 16 + ln) * HH;
#pragma unroll
            for (int kc = 0; kc < 4; ++kc) {
                v16h a = load_afrag(ar, kc * 32, laneHi);
                acc = WMMA(a, fcB[kc], acc);
            }
            _Float16* pw = predb + (size_t)(t & 1) * BB * DD;
#pragma unroll
            for (int r = 0; r < 8; ++r) {
                int b = mt * 16 + laneHi * 8 + r;
                out[(size_t)b * (SS * DD) + (size_t)t * DD + col] = acc[r];
                pw[(size_t)b * DD + col] = (_Float16)acc[r];
            }
        }
        gridbar(bar, target);
    }
}

// ---------------- prep kernels ----------------
__global__ void cvt_f32_f16(const float* __restrict__ s, _Float16* __restrict__ d, int n) {
    int i = blockIdx.x * blockDim.x + threadIdx.x;
    if (i < n) d[i] = (_Float16)s[i];
}

// combined weight [512][Kin+128] f16 = [Wih | Whh], and bias sum [512] f32
__global__ void pack_cell_w(const float* __restrict__ Wih, const float* __restrict__ Whh,
                            const float* __restrict__ bih, const float* __restrict__ bhh,
                            _Float16* __restrict__ Wc, float* __restrict__ bsum, int Kin) {
    const int K = Kin + HH;
    const int total = 512 * K;
    for (int i = blockIdx.x * blockDim.x + threadIdx.x; i < total; i += gridDim.x * blockDim.x) {
        int r = i / K, k = i - r * K;
        float v = (k < Kin) ? Wih[(size_t)r * Kin + k] : Whh[(size_t)r * HH + (k - Kin)];
        Wc[i] = (_Float16)v;
    }
    int j = blockIdx.x * blockDim.x + threadIdx.x;
    if (j < 512) bsum[j] = bih[j] + bhh[j];
}

// ---------------- host launch ----------------
extern "C" void kernel_launch(void* const* d_in, const int* in_sizes, int n_in,
                              void* d_out, int out_size, void* d_ws, size_t ws_size,
                              hipStream_t stream) {
    (void)in_sizes; (void)n_in; (void)out_size; (void)ws_size;
    const float* x        = (const float*)d_in[0];
    const float* eWih0    = (const float*)d_in[1];
    const float* eWhh0    = (const float*)d_in[2];
    const float* ebih0    = (const float*)d_in[3];
    const float* ebhh0    = (const float*)d_in[4];
    const float* eWih1    = (const float*)d_in[5];
    const float* eWhh1    = (const float*)d_in[6];
    const float* ebih1    = (const float*)d_in[7];
    const float* ebhh1    = (const float*)d_in[8];
    const float* dWih0    = (const float*)d_in[9];
    const float* dWhh0    = (const float*)d_in[10];
    const float* dbih0    = (const float*)d_in[11];
    const float* dbhh0    = (const float*)d_in[12];
    const float* dWih1    = (const float*)d_in[13];
    const float* dWhh1    = (const float*)d_in[14];
    const float* dbih1    = (const float*)d_in[15];
    const float* dbhh1    = (const float*)d_in[16];
    const float* fcW      = (const float*)d_in[17];
    const float* fcb      = (const float*)d_in[18];
    float* out            = (float*)d_out;

    char* ws = (char*)d_ws;
    size_t off = 0;
    auto carve = [&](size_t bytes) -> char* {
        off = (off + 255) & ~(size_t)255;
        char* p = ws + off;
        off += bytes;
        return p;
    };
    unsigned* bar    = (unsigned*)carve(256);
    _Float16* xh     = (_Float16*)carve((size_t)BB * SS * DD * 2);   // 16 MB
    _Float16* y0     = (_Float16*)carve((size_t)SS * BB * HH * 2);   // 32 MB
    _Float16* h0buf  = (_Float16*)carve((size_t)2 * BH * 2);
    _Float16* h1buf  = (_Float16*)carve((size_t)2 * BH * 2);
    _Float16* predb  = (_Float16*)carve((size_t)2 * BB * DD * 2);
    _Float16* w_enc0 = (_Float16*)carve((size_t)512 * 192 * 2);
    _Float16* w_enc1 = (_Float16*)carve((size_t)512 * 256 * 2);
    _Float16* w_dec0 = (_Float16*)carve((size_t)512 * 192 * 2);
    _Float16* w_dec1 = (_Float16*)carve((size_t)512 * 256 * 2);
    _Float16* fcWh   = (_Float16*)carve((size_t)DD * HH * 2);
    float*    b_enc0 = (float*)carve(512 * 4);
    float*    b_enc1 = (float*)carve(512 * 4);
    float*    b_dec0 = (float*)carve(512 * 4);
    float*    b_dec1 = (float*)carve(512 * 4);

    hipMemsetAsync(bar, 0, 256, stream);

    const int nX = BB * SS * DD;
    cvt_f32_f16<<<(nX + 255) / 256, 256, 0, stream>>>(x, xh, nX);
    cvt_f32_f16<<<(DD * HH + 255) / 256, 256, 0, stream>>>(fcW, fcWh, DD * HH);
    pack_cell_w<<<512, 256, 0, stream>>>(eWih0, eWhh0, ebih0, ebhh0, w_enc0, b_enc0, DD);
    pack_cell_w<<<512, 256, 0, stream>>>(eWih1, eWhh1, ebih1, ebhh1, w_enc1, b_enc1, HH);
    pack_cell_w<<<512, 256, 0, stream>>>(dWih0, dWhh0, dbih0, dbhh0, w_dec0, b_dec0, DD);
    pack_cell_w<<<512, 256, 0, stream>>>(dWih1, dWhh1, dbih1, dbhh1, w_dec1, b_dec1, HH);

    lstm_ae_persistent<<<NWG, 256, 0, stream>>>(
        xh, y0, h0buf, h1buf, predb,
        w_enc0, w_enc1, w_dec0, w_dec1,
        b_enc0, b_enc1, b_dec0, b_dec1,
        fcWh, fcb, out, bar);
}
